// VectorQuantizer_84980222919420
// MI455X (gfx1250) — matrix operand order repaired and checked by reading, not compiled, not run on hardware
//
#include <hip/hip_runtime.h>

// VQ-VAE vector quantizer, MI455X (gfx1250, wave32, WMMA + TDM).
//   dist = ||e_k||^2 - 2 z.e_k   (||z||^2 constant per row; irrelevant to argmin)
// Matmul: v_wmma_f32_16x16x32_f16 (f16 in, f32 accumulate).
// Codebook staged into LDS once per workgroup by the Tensor Data Mover with
// 16B row padding (144B stride = 36 banks) for conflict-free ds_load_b128.
// A volatile asm that takes the LDS pointer as an INPUT (escape) with a
// "memory" clobber makes the TDM's write visible to alias analysis while
// keeping the loads rooted at the addrspace(3) global (so they stay ds_load,
// not flat_load).
//
// Outputs concatenated flat in d_out (float):
//   [0, N*64)            zq_st (= emb[idx], straight-through forward value)
//   [N*64, N*64+N)       indices (stored as float values)
//   [N*64+N]             loss = 1.5 * mean((zq - z)^2)

typedef __attribute__((ext_vector_type(16))) _Float16     v16h;
typedef __attribute__((ext_vector_type(8)))  _Float16     v8h;
typedef __attribute__((ext_vector_type(8)))  float        v8f;
typedef __attribute__((ext_vector_type(4)))  float        f4;
typedef __attribute__((ext_vector_type(4)))  unsigned int v4u;
typedef __attribute__((ext_vector_type(8)))  int          v8i;
typedef __attribute__((ext_vector_type(4)))  int          v4i;

#define N_PTS   262144
#define DIMS    64
#define KCB     1024
#define LDS_ROW 144                    // 128B of f16 data + 16B TDM pad
#define LDS_BYTES (KCB * LDS_ROW)      // 147456 B < 160 KB

// ---------------------------------------------------------------------------
// Pass 1: emb f32 -> f16 (row-major K x 64), ||e_k||^2, zero the loss slot.
// ---------------------------------------------------------------------------
__global__ void vq_prep_kernel(const float* __restrict__ emb,
                               _Float16* __restrict__ emb_h,
                               float* __restrict__ enorm,
                               float* __restrict__ loss_out) {
    int k = blockIdx.x * blockDim.x + threadIdx.x;
    if (k == 0) *loss_out = 0.0f;      // atomics accumulate into this later
    if (k < KCB) {
        float s = 0.0f;
        #pragma unroll 8
        for (int d = 0; d < DIMS; ++d) {
            float v = emb[k * DIMS + d];
            s += v * v;
            emb_h[k * DIMS + d] = (_Float16)v;
        }
        enorm[k] = s;
    }
}

// ---------------------------------------------------------------------------
// TDM: DMA the whole f16 codebook (1024 rows x 128B) into LDS with a 16B pad
// after every 128B row (pad_interval = 32 DWORDs -> code 4, pad_amount =
// 4 DWORDs -> code 3).  D# built per CDNA5 ISA 8.3/8.4.  This toolchain
// exposes the 6-arg builtin: (g0 v4u, g1 v8i, g2 v4i, g3 v4i, v8i, cpol).
// ---------------------------------------------------------------------------
__device__ static inline void tdm_stage_codebook(const _Float16* src) {
    unsigned long long ga = (unsigned long long)src;

    v4u g0;
    g0[0] = 1u;                                          // count=1 (valid, user)
    g0[1] = 0u;                                          // lds_addr = 0 (sole LDS alloc)
    g0[2] = (unsigned int)(ga & 0xFFFFFFFFu);            // global_addr[31:0]
    g0[3] = (unsigned int)((ga >> 32) & 0x1FFFFFFu)      // global_addr[56:32]
            | (2u << 30);                                // type = 2 ("image")

    v8i g1;
    g1[0] = (int)((1u << 16)                             // data_size = 2B
                  | (1u << 20)                           // pad_enable
                  | (4u << 22)                           // pad_interval: 32 DWORDs
                  | (3u << 25));                         // pad_amount: 4 DWORDs
    g1[1] = (int)(((unsigned)DIMS & 0xFFFFu) << 16);     // tensor_dim0[15:0]=64
    g1[2] = (int)(((unsigned)KCB & 0xFFFFu) << 16);      // tdim0 hi=0 | tensor_dim1 lo=1024
    g1[3] = (int)(((unsigned)DIMS) << 16);               // tdim1 hi=0 | tile_dim0=64
    g1[4] = (int)((unsigned)KCB);                        // tile_dim1=1024 | tile_dim2=0
    g1[5] = (int)DIMS;                                   // tensor_dim0_stride = 64
    g1[6] = 0;                                           // stride0 hi | stride1 lo
    g1[7] = (int)((unsigned)(DIMS * KCB) >> 16);         // stride1 hi (unused for 2D)

    v4i zg4 = {0, 0, 0, 0};                              // groups 2/3: no dims 2..4
    v8i zg8 = {0, 0, 0, 0, 0, 0, 0, 0};
    __builtin_amdgcn_tensor_load_to_lds(g0, g1, zg4, zg4, zg8, 0);
}

// ---------------------------------------------------------------------------
// Pass 2: each wave owns 32 rows of z (two 16-row M-tiles sharing every B
// fragment); block = 8 waves = 256 rows; full-K argmin, gather, loss.
// ---------------------------------------------------------------------------
__launch_bounds__(256)
__global__ void vq_main_kernel(const float* __restrict__ z,
                               const float* __restrict__ emb,
                               const _Float16* __restrict__ emb_h,
                               const float* __restrict__ enorm,
                               float* __restrict__ out_zq,
                               float* __restrict__ out_idx,
                               float* __restrict__ loss_out) {
    __shared__ __align__(128) unsigned char lds_emb[LDS_BYTES];

    const int lane    = threadIdx.x & 31;
    const int wave    = threadIdx.x >> 5;
    const int rowBase = (blockIdx.x * 8 + wave) * 32;
    const int hi      = lane >> 4;     // which 16-lane half
    const int lo      = lane & 15;

    // ---- kick off the codebook DMA from wave 0 (TDM ignores EXEC) --------
    if (wave == 0) tdm_stage_codebook(emb_h);

    // ---- build A fragments while the DMA is in flight ---------------------
    // A layout: lane holds row M=lo; halves 0..7  -> d = 8*hi + 0..7  (+32*frag)
    //                               halves 8..15 -> d = 8*hi + 16..23 (+32*frag)
    v16h a0[2], a1[2];
    #pragma unroll
    for (int mt = 0; mt < 2; ++mt) {
        const f4* rp4 = (const f4*)(z + (rowBase + mt * 16 + lo) * DIMS + hi * 8);
        f4 c0 = rp4[0],  c1 = rp4[1];    // frag0 lo halves
        f4 c2 = rp4[4],  c3 = rp4[5];    // frag0 hi halves
        f4 c4 = rp4[8],  c5 = rp4[9];    // frag1 lo halves
        f4 c6 = rp4[12], c7 = rp4[13];   // frag1 hi halves
        #pragma unroll
        for (int i = 0; i < 4; ++i) {
            a0[mt][i]      = (_Float16)c0[i];
            a0[mt][4 + i]  = (_Float16)c1[i];
            a0[mt][8 + i]  = (_Float16)c2[i];
            a0[mt][12 + i] = (_Float16)c3[i];
            a1[mt][i]      = (_Float16)c4[i];
            a1[mt][4 + i]  = (_Float16)c5[i];
            a1[mt][8 + i]  = (_Float16)c6[i];
            a1[mt][12 + i] = (_Float16)c7[i];
        }
    }

    if (wave == 0) __builtin_amdgcn_s_wait_tensorcnt(0);
    __syncthreads();

    // ---- make the TDM's LDS write visible to alias analysis: escape the
    // LDS pointer as an asm INPUT with a memory clobber, but keep all
    // actual accesses rooted at the addrspace(3) global so the loads stay
    // ds_load_b128 (a "+v" output pointer here degrades them to flat_load).
    asm volatile("" :: "v"(&lds_emb[0]) : "memory");

    // ---- full-K argmin over 64 column tiles, B from LDS -------------------
    float best_d[2][8];
    int   best_i[2][8];
    #pragma unroll
    for (int mt = 0; mt < 2; ++mt)
        #pragma unroll
        for (int j = 0; j < 8; ++j) { best_d[mt][j] = 3.4e38f; best_i[mt][j] = 0; }

    // B layout: lane holds column n=lo; halves 0..7 -> d = 8*hi + 0..7, halves 8..15 -> d = 8*hi + 16..23 (+32*frag)
    const unsigned char* lb = (const unsigned char*)lds_emb + hi * 16;

    for (int t = 0; t < KCB / 16; ++t) {
        const int      kcol = t * 16 + lo;
        const unsigned roff = (unsigned)kcol * LDS_ROW;
        const v8h* p0 = (const v8h*)(lb + roff);        // d in [0,32)
        const v8h* p1 = (const v8h*)(lb + roff + 64);   // d in [32,64)
        v8h x0 = p0[0], x1 = p0[2];
        v8h y0 = p1[0], y1 = p1[2];
        v16h b0 = __builtin_shufflevector(x0, x1, 0,1,2,3,4,5,6,7,8,9,10,11,12,13,14,15);
        v16h b1 = __builtin_shufflevector(y0, y1, 0,1,2,3,4,5,6,7,8,9,10,11,12,13,14,15);
        float en = enorm[kcol];

        #pragma unroll
        for (int mt = 0; mt < 2; ++mt) {
            v8f c = {};
            c = __builtin_amdgcn_wmma_f32_16x16x32_f16(false, a0[mt], false, b0,
                                                       (short)0, c, false, false);
            c = __builtin_amdgcn_wmma_f32_16x16x32_f16(false, a1[mt], false, b1,
                                                       (short)0, c, false, false);
            // C layout: VGPR j, this lane -> row M = 8*hi + j, col N = lo
            #pragma unroll
            for (int j = 0; j < 8; ++j) {
                float dist = en - 2.0f * c[j];
                if (dist < best_d[mt][j]) { best_d[mt][j] = dist; best_i[mt][j] = kcol; }
            }
        }
    }

    // ---- min-reduce across the 16 lanes of each half (idx tie-break) ------
    #pragma unroll
    for (int m = 1; m <= 8; m <<= 1) {
        #pragma unroll
        for (int mt = 0; mt < 2; ++mt)
            #pragma unroll
            for (int j = 0; j < 8; ++j) {
                float od = __shfl_xor(best_d[mt][j], m, 32);
                int   oi = __shfl_xor(best_i[mt][j], m, 32);
                if (od < best_d[mt][j] ||
                    (od == best_d[mt][j] && oi < best_i[mt][j])) {
                    best_d[mt][j] = od; best_i[mt][j] = oi;
                }
            }
    }

    // ---- epilogue: indices, gather zq_st, loss partial --------------------
    float acc = 0.0f;
    #pragma unroll
    for (int mt = 0; mt < 2; ++mt) {
        #pragma unroll
        for (int j = 0; j < 8; ++j) {
            const int idx = best_i[mt][j];           // uniform across the half
            const int row = rowBase + mt * 16 + hi * 8 + j;
            if (lo == j) out_idx[row] = (float)idx;
            // 16 lanes cover the 64-float row: 4 floats each
            f4 e  = *(const f4*)(emb + idx * DIMS + lo * 4);
            f4 zv = *(const f4*)(z  + row * DIMS + lo * 4);
            *(f4*)(out_zq + row * DIMS + lo * 4) = e;
            #pragma unroll
            for (int i = 0; i < 4; ++i) {
                float d = e[i] - zv[i];
                acc += d * d;
            }
        }
    }

    // wave-level sum then one atomic per wave; scale = 1.5 / (N*D)
    #pragma unroll
    for (int m = 1; m < 32; m <<= 1) acc += __shfl_xor(acc, m, 32);
    if (lane == 0) atomicAdd(loss_out, acc * (1.5f / 16777216.0f));
}

// ---------------------------------------------------------------------------
extern "C" void kernel_launch(void* const* d_in, const int* in_sizes, int n_in,
                              void* d_out, int out_size, void* d_ws, size_t ws_size,
                              hipStream_t stream) {
    (void)in_sizes; (void)n_in; (void)out_size; (void)ws_size;

    const float* z   = (const float*)d_in[0];   // [N, 64]
    const float* emb = (const float*)d_in[1];   // [K, 64]

    float* out      = (float*)d_out;
    float* out_zq   = out;                        // N*64
    float* out_idx  = out + N_PTS * DIMS;         // N
    float* out_loss = out + N_PTS * DIMS + N_PTS; // 1

    _Float16* emb_h = (_Float16*)d_ws;                                       // 128 KB
    float*    enorm = (float*)((char*)d_ws + KCB * DIMS * sizeof(_Float16)); // 4 KB

    vq_prep_kernel<<<KCB / 256, 256, 0, stream>>>(emb, emb_h, enorm, out_loss);
    vq_main_kernel<<<N_PTS / 256, 256, 0, stream>>>(z, emb, emb_h, enorm,
                                                    out_zq, out_idx, out_loss);
}
